// EGNN_13546326852352
// MI455X (gfx1250) — matrix operand order repaired and checked by reading, not compiled, not run on hardware
//
#include <hip/hip_runtime.h>
#include <hip/hip_bf16.h>

// ---------------- problem constants ----------------
#define NN      50000
#define EE      500000
#define SDIM    128
#define VDIM    3
#define HIDN    64
#define NDEPTH  4
#define CUT     5.0f
#define PI_F    3.14159265358979323846f
#define K1ROWS  257      // exact W1 K (256 dense via WMMA + 1 d-row via VALU rank-1)
#define N3PAD   144      // 134 padded to 9*16
#define M3OUT   134
#define WAVES_E 8
#define WAVES_N 8

typedef __attribute__((ext_vector_type(16))) _Float16 v16h;
typedef __attribute__((ext_vector_type(8)))  _Float16 v8h;
typedef __attribute__((ext_vector_type(8)))  float    v8f;
typedef __attribute__((ext_vector_type(4)))  float    v4f;

__device__ __forceinline__ float silu_f(float x) {
    return x * (1.0f / (1.0f + __expf(-x)));
}

__device__ __forceinline__ void wave_lds_fence() {
    asm volatile("s_wait_dscnt 0" ::: "memory");
}

// B fragment (32x16, f16): lane L holds column N=L%16; halfs0-7 = K 8h..8h+7, halfs8-15 = K 16+8h..16+8h+7
__device__ __forceinline__ v16h load_b_frag(const _Float16* Wt, int ldn, int lane) {
    const _Float16* p = Wt + (lane & 15) + (size_t)((lane >> 4) * 8) * ldn;
    v16h b;
#pragma unroll
    for (int i = 0; i < 8; i++) { b[i] = p[(size_t)i * ldn]; b[i + 8] = p[(size_t)(16 + i) * ldn]; }
    return b;
}

// A fragment (16x32, f16) from an LDS row-major [16][ldk] panel.
// lane<16: halfs0-7 = K k0..k0+7, halfs8-15 = K k0+16..k0+23; lane>=16: +8 shift
__device__ __forceinline__ v16h load_a_frag_lds(const _Float16* Hs, int ldk, int k0, int lane) {
    int mm = lane & 15;
    const _Float16* p = Hs + mm * ldk + k0 + ((lane < 16) ? 0 : 8);
    v8h lo = *(const v8h*)p;
    v8h hi = *(const v8h*)(p + 16);
    v16h a;
#pragma unroll
    for (int i = 0; i < 8; i++) { a[i] = lo[i]; a[i + 8] = hi[i]; }
    return a;
}

// A fragment (16x32) gathered from a per-lane fp32 row pointer (already +0/+8 shifted),
// converting fp32 -> f16 on the fly.
__device__ __forceinline__ v16h cvt_a_frag(const float* rowp) {
    v4f f0 = *(const v4f*)(rowp);
    v4f f1 = *(const v4f*)(rowp + 4);
    v4f f2 = *(const v4f*)(rowp + 16);
    v4f f3 = *(const v4f*)(rowp + 20);
    v16h a;
#pragma unroll
    for (int i = 0; i < 4; i++) {
        a[i]      = (_Float16)f0[i];
        a[4 + i]  = (_Float16)f1[i];
        a[8 + i]  = (_Float16)f2[i];
        a[12 + i] = (_Float16)f3[i];
    }
    return a;
}

#define WMMA_F16(A, B, C) \
    __builtin_amdgcn_wmma_f32_16x16x32_f16(false, (A), false, (B), (short)0, (C), false, false)

// ---------------- helper kernels ----------------
__global__ void egnn_zero_f(float* __restrict__ p, int n) {
    int i = blockIdx.x * blockDim.x + threadIdx.x;
    if (i < n) p[i] = 0.0f;
}

__global__ void egnn_init_copy(const float* __restrict__ s_in, const float* __restrict__ v_in,
                               float* __restrict__ s_cur, float* __restrict__ v_cur) {
    int i = blockIdx.x * blockDim.x + threadIdx.x;
    if (i < NN * SDIM) s_cur[i] = s_in[i];
    else if (i < NN * SDIM + NN * 9) v_cur[i - NN * SDIM] = v_in[i - NN * SDIM];
}

__global__ void egnn_count(const int* __restrict__ eidx, float* __restrict__ cnt) {
    int i = blockIdx.x * blockDim.x + threadIdx.x;
    if (i < EE) atomicAdd(&cnt[eidx[i]], 1.0f);
}

__global__ void egnn_invcnt(const float* __restrict__ cnt, float* __restrict__ icnt) {
    int i = blockIdx.x * blockDim.x + threadIdx.x;
    if (i < NN) icnt[i] = 1.0f / fmaxf(cnt[i], 1.0f);
}

__global__ void egnn_cvt_plain(const float* __restrict__ W, _Float16* __restrict__ Wp, int n) {
    int i = blockIdx.x * blockDim.x + threadIdx.x;
    if (i < n) Wp[i] = (_Float16)W[i];
}

__global__ void egnn_cvt_w3(const float* __restrict__ W3, _Float16* __restrict__ W3p) {
    int i = blockIdx.x * blockDim.x + threadIdx.x;  // over NDEPTH*HIDN*N3PAD
    if (i >= NDEPTH * HIDN * N3PAD) return;
    int l = i / (HIDN * N3PAD);
    int r = i % (HIDN * N3PAD);
    int k = r / N3PAD, n = r % N3PAD;
    W3p[i] = (n < M3OUT) ? (_Float16)W3[(l * HIDN + k) * M3OUT + n] : (_Float16)0.0f;
}

__global__ void egnn_out_copy(const float* __restrict__ s_cur, const float* __restrict__ v_cur,
                              float* __restrict__ out) {
    int i = blockIdx.x * blockDim.x + threadIdx.x;
    if (i < NN * SDIM) out[i] = s_cur[i];
    else if (i < NN * SDIM + NN * 9) out[i] = v_cur[i - NN * SDIM];
}

// ---------------- edge MLP + scatter (WMMA) ----------------
__global__ __launch_bounds__(32 * WAVES_E) void egnn_edge_kernel(
    const float* __restrict__ s, const float* __restrict__ v,
    const int* __restrict__ eidx, const float* __restrict__ dref,
    const float* __restrict__ rvec,
    const _Float16* __restrict__ W1f, const float* __restrict__ b1,
    const _Float16* __restrict__ W2f, const float* __restrict__ b2,
    const _Float16* __restrict__ W3f, const float* __restrict__ b3,
    float* __restrict__ s_agg, float* __restrict__ v_agg, int layer)
{
    __shared__ __align__(32) _Float16 sW1[K1ROWS * HIDN];      // 32.1 KB (rows 0..255 WMMA, row 256 VALU)
    __shared__ __align__(32) _Float16 sW2[HIDN * HIDN];        // 8 KB
    __shared__ __align__(32) _Float16 sW3[HIDN * N3PAD];       // 18 KB
    __shared__ __align__(32) _Float16 sH1[WAVES_E][16 * HIDN]; // 16 KB
    __shared__ __align__(32) _Float16 sH2[WAVES_E][16 * HIDN]; // 16 KB
    __shared__ float sM8[WAVES_E][16 * 8];                     // 4 KB
    __shared__ int   sDst[WAVES_E][16];
    __shared__ float sCc[WAVES_E][16];
    __shared__ float sDe[WAVES_E][16];

    const int tid = threadIdx.x;
    // stage all three weight matrices for this layer into LDS (shared by all waves)
    const _Float16* W1g = W1f + (size_t)layer * K1ROWS * HIDN;
    const _Float16* W2g = W2f + (size_t)layer * HIDN * HIDN;
    const _Float16* W3g = W3f + (size_t)layer * HIDN * N3PAD;
    for (int i = tid; i < K1ROWS * HIDN; i += blockDim.x) sW1[i] = W1g[i];
    for (int i = tid; i < HIDN * HIDN; i += blockDim.x)   sW2[i] = W2g[i];
    for (int i = tid; i < HIDN * N3PAD; i += blockDim.x)  sW3[i] = W3g[i];
    __syncthreads();

    const int wave = tid >> 5, lane = tid & 31;
    const int tile = blockIdx.x * WAVES_E + wave;
    if (tile * 16 >= EE) return;
    const int tb = tile * 16;
    const int m = lane & 15;
    const int e = tb + m;

    const int   dsti = eidx[e];
    const int   srci = eidx[EE + e];
    const float de   = dref[e];
    const float Ce   = (de < CUT) ? 0.5f * (__cosf(PI_F * de / CUT) + 1.0f) : 0.0f;
    if (lane < 16) { sDst[wave][m] = dsti; sCc[wave][m] = Ce; sDe[wave][m] = de; }
    wave_lds_fence();

    // prefetch the tail-phase random gathers (r[e], v[src]) behind the MLP
    __builtin_prefetch(rvec + (size_t)e * 3, 0, 1);
    __builtin_prefetch(v + (size_t)srci * 9, 0, 1);

    // ---- GEMM1: h1 = silu([s_dst | s_src | d] @ W1 + b1)
    //      K=0..255 dense via 8x WMMA chunks, K=256 (d column) via VALU rank-1 update
    const float* b1g = b1 + layer * HIDN;
    v8f acc[4];
#pragma unroll
    for (int n = 0; n < 4; n++) {
        float bv = b1g[n * 16 + m];
#pragma unroll
        for (int r = 0; r < 8; r++) acc[n][r] = bv;
    }
    for (int kk = 0; kk < 8; kk++) {
        const float* base = (kk < 4) ? (s + (size_t)dsti * SDIM + kk * 32)
                                     : (s + (size_t)srci * SDIM + (kk - 4) * 32);
        v16h a = cvt_a_frag(base + ((lane < 16) ? 0 : 8));
#pragma unroll
        for (int n = 0; n < 4; n++) {
            v16h b = load_b_frag(&sW1[(kk * 32) * HIDN + n * 16], HIDN, lane);
            acc[n] = WMMA_F16(a, b, acc[n]);
        }
    }
#pragma unroll
    for (int n = 0; n < 4; n++) {
        float w256 = (float)sW1[256 * HIDN + n * 16 + m];  // W1[256, col]
#pragma unroll
        for (int r = 0; r < 8; r++) {
            int row = r + ((lane >> 4) << 3);
            acc[n][r] = fmaf(sDe[wave][row], w256, acc[n][r]);
            sH1[wave][row * HIDN + n * 16 + m] = (_Float16)silu_f(acc[n][r]);
        }
    }
    wave_lds_fence();

    // ---- GEMM2: h2 = silu(h1 @ W2 + b2)   (16x64)x(64x64)
    const float* b2g = b2 + layer * HIDN;
    v8f acc2[4];
#pragma unroll
    for (int n = 0; n < 4; n++) {
        float bv = b2g[n * 16 + m];
#pragma unroll
        for (int r = 0; r < 8; r++) acc2[n][r] = bv;
    }
#pragma unroll
    for (int kk = 0; kk < 2; kk++) {
        v16h a = load_a_frag_lds(&sH1[wave][0], HIDN, kk * 32, lane);
#pragma unroll
        for (int n = 0; n < 4; n++) {
            v16h b = load_b_frag(&sW2[(kk * 32) * HIDN + n * 16], HIDN, lane);
            acc2[n] = WMMA_F16(a, b, acc2[n]);
        }
    }
#pragma unroll
    for (int n = 0; n < 4; n++)
#pragma unroll
        for (int r = 0; r < 8; r++) {
            int row = r + ((lane >> 4) << 3);
            sH2[wave][row * HIDN + n * 16 + m] = (_Float16)silu_f(acc2[n][r]);
        }
    wave_lds_fence();

    // ---- GEMM3: m = h2 @ W3 + b3  (16x64)x(64x144); tiles 0-7 = ms, tile 8 = gv/gr
    const float* b3g = b3 + layer * M3OUT;
    for (int n = 0; n < 9; n++) {
        int col = n * 16 + m;
        float bv = (col < M3OUT) ? b3g[col] : 0.0f;
        v8f acc3;
#pragma unroll
        for (int r = 0; r < 8; r++) acc3[r] = bv;
#pragma unroll
        for (int kk = 0; kk < 2; kk++) {
            v16h a = load_a_frag_lds(&sH2[wave][0], HIDN, kk * 32, lane);
            v16h b = load_b_frag(&sW3[(kk * 32) * N3PAD + n * 16], N3PAD, lane);
            acc3 = WMMA_F16(a, b, acc3);
        }
        if (n < 8) {
            // scalar message scatter: s_agg[dst] += ms * C
#pragma unroll
            for (int r = 0; r < 8; r++) {
                int row = r + ((lane >> 4) << 3);
                float val = acc3[r] * sCc[wave][row];
                atomicAdd(&s_agg[(size_t)sDst[wave][row] * SDIM + n * 16 + m], val);
            }
        } else {
            if (m < 6) {
#pragma unroll
                for (int r = 0; r < 8; r++) {
                    int row = r + ((lane >> 4) << 3);
                    sM8[wave][row * 8 + m] = acc3[r];
                }
            }
        }
    }
    wave_lds_fence();

    // ---- vector message: vm = (v[src]*gv + r*gr) * C, scatter to v_agg
    if (lane < 16) {
        float g[3], h[3];
#pragma unroll
        for (int c = 0; c < 3; c++) { g[c] = sM8[wave][m * 8 + c]; h[c] = sM8[wave][m * 8 + 3 + c]; }
        const float* vs = v + (size_t)srci * 9;
        const float* re = rvec + (size_t)e * 3;
        float*       va = v_agg + (size_t)dsti * 9;
        float r0 = re[0], r1 = re[1], r2v = re[2];
#pragma unroll
        for (int c = 0; c < 3; c++) {
            atomicAdd(&va[c * 3 + 0], (vs[c * 3 + 0] * g[c] + r0  * h[c]) * Ce);
            atomicAdd(&va[c * 3 + 1], (vs[c * 3 + 1] * g[c] + r1  * h[c]) * Ce);
            atomicAdd(&va[c * 3 + 2], (vs[c * 3 + 2] * g[c] + r2v * h[c]) * Ce);
        }
    }
}

// ---------------- node update (WMMA) ----------------
__global__ __launch_bounds__(32 * WAVES_N) void egnn_node_kernel(
    float* __restrict__ s, float* __restrict__ v,
    const float* __restrict__ s_agg, const float* __restrict__ v_agg,
    const float* __restrict__ icnt,
    const _Float16* __restrict__ Wn1f, const float* __restrict__ bn1,
    const _Float16* __restrict__ Wn2f, const float* __restrict__ bn2, int layer)
{
    __shared__ __align__(32) _Float16 sWn1[2 * SDIM * HIDN];   // 32 KB
    __shared__ __align__(32) _Float16 sWn2[HIDN * SDIM];       // 16 KB
    __shared__ __align__(32) _Float16 sU[WAVES_N][16 * HIDN];  // 16 KB

    const int tid = threadIdx.x;
    const _Float16* W1g = Wn1f + (size_t)layer * 2 * SDIM * HIDN;
    const _Float16* W2g = Wn2f + (size_t)layer * HIDN * SDIM;
    for (int i = tid; i < 2 * SDIM * HIDN; i += blockDim.x) sWn1[i] = W1g[i];
    for (int i = tid; i < HIDN * SDIM; i += blockDim.x)     sWn2[i] = W2g[i];
    __syncthreads();

    const int wave = tid >> 5, lane = tid & 31;
    const int tile = blockIdx.x * WAVES_N + wave;
    if (tile * 16 >= NN) return;
    const int tb = tile * 16;
    const int m = lane & 15;

    // ---- u = silu([s | s_agg] @ Wn1 + bn1)   (16x256)x(256x64)
    const float* bn1g = bn1 + layer * HIDN;
    v8f acc[4];
#pragma unroll
    for (int n = 0; n < 4; n++) {
        float bv = bn1g[n * 16 + m];
#pragma unroll
        for (int r = 0; r < 8; r++) acc[n][r] = bv;
    }
    for (int kk = 0; kk < 8; kk++) {
        const float* base = (kk < 4) ? (s + (size_t)(tb + m) * SDIM + kk * 32)
                                     : (s_agg + (size_t)(tb + m) * SDIM + (kk - 4) * 32);
        v16h a = cvt_a_frag(base + ((lane < 16) ? 0 : 8));
#pragma unroll
        for (int n = 0; n < 4; n++) {
            v16h b = load_b_frag(&sWn1[(kk * 32) * HIDN + n * 16], HIDN, lane);
            acc[n] = WMMA_F16(a, b, acc[n]);
        }
    }
#pragma unroll
    for (int n = 0; n < 4; n++)
#pragma unroll
        for (int r = 0; r < 8; r++) {
            int row = r + ((lane >> 4) << 3);
            sU[wave][row * HIDN + n * 16 + m] = (_Float16)silu_f(acc[n][r]);
        }
    wave_lds_fence();

    // ---- s = s + u @ Wn2 + bn2    (16x64)x(64x128), accumulate into loaded s
    const float* bn2g = bn2 + layer * SDIM;
    for (int n = 0; n < 8; n++) {
        int col = n * 16 + m;
        float bv = bn2g[col];
        v8f acc2;
#pragma unroll
        for (int r = 0; r < 8; r++) {
            int row = r + ((lane >> 4) << 3);
            acc2[r] = s[(size_t)(tb + row) * SDIM + col] + bv;
        }
#pragma unroll
        for (int kk = 0; kk < 2; kk++) {
            v16h a = load_a_frag_lds(&sU[wave][0], HIDN, kk * 32, lane);
            v16h b = load_b_frag(&sWn2[(kk * 32) * SDIM + n * 16], SDIM, lane);
            acc2 = WMMA_F16(a, b, acc2);
        }
#pragma unroll
        for (int r = 0; r < 8; r++) {
            int row = r + ((lane >> 4) << 3);
            s[(size_t)(tb + row) * SDIM + col] = acc2[r];
        }
    }

    // ---- v = v + v_agg * inv_cnt
    if (lane < 16) {
        int node = tb + m;
        float ic = icnt[node];
#pragma unroll
        for (int j = 0; j < 9; j++)
            v[(size_t)node * 9 + j] += v_agg[(size_t)node * 9 + j] * ic;
    }
}

// ---------------- launcher ----------------
extern "C" void kernel_launch(void* const* d_in, const int* in_sizes, int n_in,
                              void* d_out, int out_size, void* d_ws, size_t ws_size,
                              hipStream_t stream) {
    (void)in_sizes; (void)n_in; (void)out_size; (void)ws_size;

    const float* s_in = (const float*)d_in[0];
    const float* v_in = (const float*)d_in[1];
    const int*   eidx = (const int*)d_in[2];
    const float* dref = (const float*)d_in[3];
    const float* rvec = (const float*)d_in[4];
    const float* W1   = (const float*)d_in[5];
    const float* b1   = (const float*)d_in[6];
    const float* W2   = (const float*)d_in[7];
    const float* b2   = (const float*)d_in[8];
    const float* W3   = (const float*)d_in[9];
    const float* b3   = (const float*)d_in[10];
    const float* Wn1  = (const float*)d_in[11];
    const float* bn1  = (const float*)d_in[12];
    const float* Wn2  = (const float*)d_in[13];
    const float* bn2  = (const float*)d_in[14];

    char* ws = (char*)d_ws;
    size_t off = 0;
    auto take = [&](size_t bytes) -> char* {
        char* p = ws + off;
        off += (bytes + 255) & ~(size_t)255;
        return p;
    };
    float* s_cur = (float*)take((size_t)NN * SDIM * 4);
    float* v_cur = (float*)take((size_t)NN * 9 * 4);
    float* s_agg = (float*)take((size_t)NN * SDIM * 4);
    float* v_agg = (float*)take((size_t)NN * 9 * 4);
    float* cnt   = (float*)take((size_t)NN * 4);
    float* icnt  = (float*)take((size_t)NN * 4);
    _Float16* W1p  = (_Float16*)take((size_t)NDEPTH * K1ROWS * HIDN * 2);
    _Float16* W2p  = (_Float16*)take((size_t)NDEPTH * HIDN * HIDN * 2);
    _Float16* W3p  = (_Float16*)take((size_t)NDEPTH * HIDN * N3PAD * 2);
    _Float16* Wn1p = (_Float16*)take((size_t)NDEPTH * 2 * SDIM * HIDN * 2);
    _Float16* Wn2p = (_Float16*)take((size_t)NDEPTH * HIDN * SDIM * 2);

    const int T = 256;
    auto blk = [](int n, int t) { return (n + t - 1) / t; };

    // weight conversion (f32 -> f16; only W3 needs N padding)
    egnn_cvt_plain<<<blk(NDEPTH * K1ROWS * HIDN, T), T, 0, stream>>>(W1, W1p, NDEPTH * K1ROWS * HIDN);
    egnn_cvt_plain<<<blk(NDEPTH * HIDN * HIDN, T), T, 0, stream>>>(W2, W2p, NDEPTH * HIDN * HIDN);
    egnn_cvt_w3<<<blk(NDEPTH * HIDN * N3PAD, T), T, 0, stream>>>(W3, W3p);
    egnn_cvt_plain<<<blk(NDEPTH * 2 * SDIM * HIDN, T), T, 0, stream>>>(Wn1, Wn1p, NDEPTH * 2 * SDIM * HIDN);
    egnn_cvt_plain<<<blk(NDEPTH * HIDN * SDIM, T), T, 0, stream>>>(Wn2, Wn2p, NDEPTH * HIDN * SDIM);

    // state init + degree counts
    egnn_init_copy<<<blk(NN * SDIM + NN * 9, T), T, 0, stream>>>(s_in, v_in, s_cur, v_cur);
    egnn_zero_f<<<blk(NN, T), T, 0, stream>>>(cnt, NN);
    egnn_count<<<blk(EE, T), T, 0, stream>>>(eidx, cnt);
    egnn_invcnt<<<blk(NN, T), T, 0, stream>>>(cnt, icnt);

    const int edge_tiles = EE / 16;                 // 31250
    const int node_tiles = NN / 16;                 // 3125
    const int edge_blocks = blk(edge_tiles, WAVES_E);
    const int node_blocks = blk(node_tiles, WAVES_N);

    for (int l = 0; l < NDEPTH; l++) {
        egnn_zero_f<<<blk(NN * SDIM, T), T, 0, stream>>>(s_agg, NN * SDIM);
        egnn_zero_f<<<blk(NN * 9, T), T, 0, stream>>>(v_agg, NN * 9);
        egnn_edge_kernel<<<edge_blocks, 32 * WAVES_E, 0, stream>>>(
            s_cur, v_cur, eidx, dref, rvec,
            W1p, b1, W2p, b2, W3p, b3, s_agg, v_agg, l);
        egnn_node_kernel<<<node_blocks, 32 * WAVES_N, 0, stream>>>(
            s_cur, v_cur, s_agg, v_agg, icnt,
            Wn1p, bn1, Wn2p, bn2, l);
    }

    egnn_out_copy<<<blk(NN * SDIM + NN * 9, T), T, 0, stream>>>(s_cur, v_cur, (float*)d_out);
}
